// ConvAttention_2302102470963
// MI455X (gfx1250) — compile-verified, matrix operand order checked by reading, not run on hardware
//
#include <hip/hip_runtime.h>
#include <hip/hip_bf16.h>

typedef __attribute__((ext_vector_type(16))) __bf16 v16bf;
typedef __attribute__((ext_vector_type(8)))  __bf16 v8bf;
typedef __attribute__((ext_vector_type(8)))  float  v8f;

#define HH 16
#define SS 2048
#define DD 128
#define TQ 128           // q rows per workgroup (8 waves x 16)
#define TK 128           // k cols per inner step
#define RS 168           // sraw row stride (bf16 elems), 336B -> good bank spread
#define KST 136          // sk / svt row stride (bf16 elems), 272B -> 16B aligned

__device__ __forceinline__ v8f fzero8() {
  v8f z;
#pragma unroll
  for (int i = 0; i < 8; ++i) z[i] = 0.f;
  return z;
}

// Load a 16x32 B-fragment slice for one lane from bf16 LDS:
// lane holds col n = lane%16 (the row pointer passed in), elements follow the
// 16-bit WMMA K pattern: K = kk + i + 8*hf + (i>=8 ? 8 : 0)  -> two 16B segments.
__device__ __forceinline__ v16bf ldfrag(const __bf16* row, int kk, int hf) {
  v8bf lo = *(const v8bf*)(row + kk + 8 * hf);
  v8bf hi = *(const v8bf*)(row + kk + 16 + 8 * hf);
  return __builtin_shufflevector(lo, hi, 0, 1, 2, 3, 4, 5, 6, 7,
                                 8, 9, 10, 11, 12, 13, 14, 15);
}

__device__ __forceinline__ v8f wmma_bf16(v16bf a, v16bf b, v8f c) {
  return __builtin_amdgcn_wmma_f32_16x16x32_bf16(false, a, false, b,
                                                 (short)0, c, false, false);
}

__global__ __launch_bounds__(256) void conv_attn_flash(
    const float* __restrict__ Q, const float* __restrict__ K,
    const float* __restrict__ V, const float* __restrict__ CW,
    float* __restrict__ Out, float* __restrict__ Mo, float* __restrict__ Lo) {
  // raw (masked,scaled) scores: 9 strips x 16 rows, 10 col tiles x 16
  __shared__ __bf16 sraw[9 * 16 * RS];        // 48384 B
  __shared__ __bf16 sk[(TK + 32) * KST];      // 43520 B  K tile (rows kc0-16..kc0+143)
  __shared__ __bf16 svt[DD * KST];            // 34816 B  V^T tile (D x TK)

  const int tid  = threadIdx.x;
  const int w    = tid >> 5;
  const int lane = tid & 31;
  const int m    = lane & 15;   // row-within-tile for A/B frags, col for C frags
  const int hf   = lane >> 4;   // half-wave
  const int head = blockIdx.y;
  const int bx   = blockIdx.x;
  const int q0   = bx * TQ;
  const float scale = 0.08838834764831845f;   // 1/sqrt(128)

  // per-head 3x3 conv weights (uniform -> scalar loads)
  const float* wp = CW + head * 9;
  const float w00 = wp[0], w01 = wp[1], w02 = wp[2];
  const float w10 = wp[3], w11 = wp[4], w12 = wp[5];
  const float w20 = wp[6], w21 = wp[7], w22 = wp[8];

  // Q A-fragments: own strip (rows q0+16w..+15) and halo strip (rows q0-16..-1)
  v16bf aq[4], aqh[4];
  {
    const int r_own = q0 + 16 * w + m;
    const int r_hal = max(q0 - 16 + m, 0);   // clamp; masked to 0 at raw store
    const float* qr0 = Q + ((size_t)head * SS + r_own) * DD;
    const float* qr1 = Q + ((size_t)head * SS + r_hal) * DD;
#pragma unroll
    for (int c = 0; c < 4; ++c) {
#pragma unroll
      for (int i = 0; i < 8; ++i) {
        aq [c][i]     = (__bf16)qr0[32 * c + 8 * hf + i];
        aq [c][8 + i] = (__bf16)qr0[32 * c + 16 + 8 * hf + i];
        aqh[c][i]     = (__bf16)qr1[32 * c + 8 * hf + i];
        aqh[c][8 + i] = (__bf16)qr1[32 * c + 16 + 8 * hf + i];
      }
    }
  }

  v8f acc[8];
#pragma unroll
  for (int n = 0; n < 8; ++n) acc[n] = fzero8();
  float mrun = -__builtin_inff();
  float lrun = 0.f;
  const int qg = q0 + 16 * w + m;   // this lane's output row (softmax row owner)

  for (int t = 0; t <= bx; ++t) {
    const int kc0 = t * TK;

    // ---- stage K tile (bf16) : rows kc0-16 .. kc0+143 ----
    for (int idx = tid; idx < (TK + 32) * (DD / 4); idx += 256) {
      const int r = idx >> 5, seg = idx & 31;
      const int kr = min(max(kc0 - 16 + r, 0), SS - 1);
      const float4 f = *(const float4*)(K + ((size_t)head * SS + kr) * DD + seg * 4);
      __bf16* d = &sk[r * KST + seg * 4];
      d[0] = (__bf16)f.x; d[1] = (__bf16)f.y; d[2] = (__bf16)f.z; d[3] = (__bf16)f.w;
    }
    // ---- stage V^T tile (bf16) ----
    for (int idx = tid; idx < TK * (DD / 4); idx += 256) {
      const int r = idx >> 5, seg = idx & 31;
      const float4 f = *(const float4*)(V + ((size_t)head * SS + kc0 + r) * DD + seg * 4);
      svt[(seg * 4 + 0) * KST + r] = (__bf16)f.x;
      svt[(seg * 4 + 1) * KST + r] = (__bf16)f.y;
      svt[(seg * 4 + 2) * KST + r] = (__bf16)f.z;
      svt[(seg * 4 + 3) * KST + r] = (__bf16)f.w;
    }
    // prefetch next step's K/V tiles into L2 (global_prefetch_b8)
    if (t < bx) {
      const int nk = kc0 + TK;
      if (tid < TK + 32) {
        const int r = min(max(nk - 16 + tid, 0), SS - 1);
        __builtin_prefetch(K + ((size_t)head * SS + r) * DD, 0, 1);
      } else {
        const int r = tid - (TK + 32);   // 0..95
        __builtin_prefetch(V + ((size_t)head * SS + nk + r) * DD, 0, 1);
        if (r < 32)
          __builtin_prefetch(V + ((size_t)head * SS + nk + 96 + r) * DD, 0, 1);
      }
    }
    __syncthreads();

    // ---- QK^T raw score tiles (masked, scaled) -> sraw ----
    // Own strip: all 10 column tiles.
    {
      const int r0 = q0 + 16 * w;
      for (int ct = 0; ct < 10; ++ct) {
        const int ccol  = 16 * ct + m;
        const int cglob = kc0 - 16 + 16 * ct + m;
        const bool live = (kc0 - 16 + 16 * ct) <= (r0 + 15);   // wave-uniform
        v8f c = fzero8();
        if (live) {
          const __bf16* br = &sk[(16 * ct + m) * KST];
          c = wmma_bf16(aq[0], ldfrag(br, 0, hf), c);
          c = wmma_bf16(aq[1], ldfrag(br, 32, hf), c);
          c = wmma_bf16(aq[2], ldfrag(br, 64, hf), c);
          c = wmma_bf16(aq[3], ldfrag(br, 96, hf), c);
        }
#pragma unroll
        for (int j = 0; j < 8; ++j) {
          const int rg = r0 + j + 8 * hf;
          float v = 0.f;
          if (live && cglob >= 0 && cglob <= rg) v = c[j] * scale;
          sraw[(16 * (w + 1) + j + 8 * hf) * RS + ccol] = (__bf16)v;
        }
      }
    }
    // Halo strip (rows q0-16..q0-1): its 10 col tiles spread across the 8 waves.
    {
      const int r0 = q0 - 16;
      for (int ct = w; ct < 10; ct += 8) {
        const int ccol  = 16 * ct + m;
        const int cglob = kc0 - 16 + 16 * ct + m;
        const bool live = (kc0 - 16 + 16 * ct) <= (r0 + 15);   // wave-uniform
        v8f c = fzero8();
        if (live) {
          const __bf16* br = &sk[(16 * ct + m) * KST];
          c = wmma_bf16(aqh[0], ldfrag(br, 0, hf), c);
          c = wmma_bf16(aqh[1], ldfrag(br, 32, hf), c);
          c = wmma_bf16(aqh[2], ldfrag(br, 64, hf), c);
          c = wmma_bf16(aqh[3], ldfrag(br, 96, hf), c);
        }
#pragma unroll
        for (int j = 0; j < 8; ++j) {
          const int rg = r0 + j + 8 * hf;
          float v = 0.f;
          if (live && rg >= 0 && cglob >= 0 && cglob <= rg) v = c[j] * scale;
          sraw[(j + 8 * hf) * RS + ccol] = (__bf16)v;
        }
      }
    }
    __syncthreads();

    // ---- 3x3 causal conv -> masked scores, directly in P A-fragment layout ----
    v16bf pf[4];
    float tmax = -__builtin_inff();
    const int rr = 16 + 16 * w + m;   // sraw row of this lane's q row
#pragma unroll
    for (int kc = 0; kc < 4; ++kc) {
#pragma unroll
      for (int h8 = 0; h8 < 2; ++h8) {
        const int cb = 32 * kc + 16 * h8 + 8 * hf;   // block of 8 k-locals
        // need raw cols (15+cb .. 24+cb); fetch aligned 16B windows [8+cb..31+cb]
        const __bf16* b2 = &sraw[(rr - 2) * RS + 8 + cb];
        const __bf16* b1 = &sraw[(rr - 1) * RS + 8 + cb];
        const __bf16* b0 = &sraw[(rr    ) * RS + 8 + cb];
        const v8bf r2a = *(const v8bf*)(b2),  r2b = *(const v8bf*)(b2 + 8),
                   r2c = *(const v8bf*)(b2 + 16);
        const v8bf r1a = *(const v8bf*)(b1),  r1b = *(const v8bf*)(b1 + 8),
                   r1c = *(const v8bf*)(b1 + 16);
        const v8bf r0a = *(const v8bf*)(b0),  r0b = *(const v8bf*)(b0 + 8),
                   r0c = *(const v8bf*)(b0 + 16);
        float t2[10], t1[10], t0[10];
        t2[0] = (float)r2a[7]; t1[0] = (float)r1a[7]; t0[0] = (float)r0a[7];
#pragma unroll
        for (int u = 0; u < 8; ++u) {
          t2[1 + u] = (float)r2b[u];
          t1[1 + u] = (float)r1b[u];
          t0[1 + u] = (float)r0b[u];
        }
        t2[9] = (float)r2c[0]; t1[9] = (float)r1c[0]; t0[9] = (float)r0c[0];
#pragma unroll
        for (int k = 0; k < 8; ++k) {
          const float o = w00 * t2[k] + w01 * t2[k + 1] + w02 * t2[k + 2]
                        + w10 * t1[k] + w11 * t1[k + 1] + w12 * t1[k + 2]
                        + w20 * t0[k] + w21 * t0[k + 1] + w22 * t0[k + 2];
          const int kgl = kc0 + cb + k;
          const float csv = (kgl <= qg) ? o : -__builtin_inff();
          tmax = fmaxf(tmax, csv);
          pf[kc][8 * h8 + k] = (__bf16)csv;   // stash pre-softmax value
        }
      }
    }
    // ---- online softmax (two lanes per row -> single xor-16 exchange) ----
    tmax = fmaxf(tmax, __shfl_xor(tmax, 16));
    const float mnew  = fmaxf(mrun, tmax);
    const float alpha = __expf(mrun - mnew);   // exp(-inf)=0 on first tile
    float psum = 0.f;
#pragma unroll
    for (int kc = 0; kc < 4; ++kc)
#pragma unroll
      for (int i = 0; i < 16; ++i) {
        const float pv = __expf((float)pf[kc][i] - mnew);
        psum += pv;
        pf[kc][i] = (__bf16)pv;
      }
    psum += __shfl_xor(psum, 16);
    lrun = lrun * alpha + psum;
    mrun = mnew;
    // rescale accumulators: C-element j lives on row j+8*hf (hoist 8 broadcasts)
    float al[8];
#pragma unroll
    for (int j = 0; j < 8; ++j) al[j] = __shfl(alpha, j + 8 * hf);
#pragma unroll
    for (int n = 0; n < 8; ++n)
#pragma unroll
      for (int j = 0; j < 8; ++j)
        acc[n][j] *= al[j];

    // ---- P @ V : 8 D-tiles x 4 K-chunks ----
#pragma unroll
    for (int n = 0; n < 8; ++n) {
      const __bf16* vr = &svt[(16 * n + m) * KST];
      acc[n] = wmma_bf16(pf[0], ldfrag(vr, 0, hf), acc[n]);
      acc[n] = wmma_bf16(pf[1], ldfrag(vr, 32, hf), acc[n]);
      acc[n] = wmma_bf16(pf[2], ldfrag(vr, 64, hf), acc[n]);
      acc[n] = wmma_bf16(pf[3], ldfrag(vr, 96, hf), acc[n]);
    }
    __syncthreads();   // before next step overwrites sk/svt/sraw
  }

  // ---- epilogue: normalize, store O, M, L ----
  const float invl = 1.f / lrun;
  float il[8];
#pragma unroll
  for (int j = 0; j < 8; ++j) il[j] = __shfl(invl, j + 8 * hf);
#pragma unroll
  for (int n = 0; n < 8; ++n) {
#pragma unroll
    for (int j = 0; j < 8; ++j) {
      const int row = q0 + 16 * w + j + 8 * hf;
      Out[((size_t)head * SS + row) * DD + 16 * n + m] = acc[n][j] * il[j];
    }
  }
  if (lane < 16) {
    Mo[(size_t)head * SS + q0 + 16 * w + lane] = mrun;
    Lo[(size_t)head * SS + q0 + 16 * w + lane] = lrun;
  }
}

extern "C" void kernel_launch(void* const* d_in, const int* in_sizes, int n_in,
                              void* d_out, int out_size, void* d_ws, size_t ws_size,
                              hipStream_t stream) {
  (void)in_sizes; (void)n_in; (void)out_size; (void)d_ws; (void)ws_size;
  const float* Q  = (const float*)d_in[0];
  const float* K  = (const float*)d_in[1];
  const float* V  = (const float*)d_in[2];
  const float* CW = (const float*)d_in[3];
  float* Out = (float*)d_out;
  float* Mo  = Out + (size_t)HH * SS * DD;
  float* Lo  = Mo + (size_t)HH * SS;
  dim3 grid(SS / TQ, HH);   // (16 q-tiles, 16 heads)
  dim3 block(256);          // 8 waves (wave32)
  conv_attn_flash<<<grid, block, 0, stream>>>(Q, K, V, CW, Out, Mo, Lo);
}
